// HierarchicalRegimeVQVAE_80607946211620
// MI455X (gfx1250) — matrix operand order repaired and checked by reading, not compiled, not run on hardware
//
#include <hip/hip_runtime.h>

// ---------------------------------------------------------------------------
// Types
// ---------------------------------------------------------------------------
typedef __bf16 bf16_t;
typedef __attribute__((ext_vector_type(16))) __bf16    v16bf;
typedef __attribute__((ext_vector_type(8)))  float     v8f;
typedef __attribute__((ext_vector_type(4)))  unsigned  u32x4;
typedef __attribute__((ext_vector_type(2)))  unsigned  u32x2;
typedef __attribute__((ext_vector_type(4)))  float     f32x4;

struct FragBits { u32x4 lo, hi; };

__device__ __forceinline__ unsigned short f2bf_bits(float f) {
  unsigned u = __builtin_bit_cast(unsigned, f);
  u += 0x7FFFu + ((u >> 16) & 1u);            // round-to-nearest-even
  return (unsigned short)(u >> 16);
}
__device__ __forceinline__ bf16_t f2bf(float f) {
  unsigned short h = f2bf_bits(f);
  return __builtin_bit_cast(bf16_t, h);
}

// ---------------------------------------------------------------------------
// WMMA fragment loaders (CDNA5 wave32 layouts, 16-bit operands)
// A (16x32, MxK): lane L<16 -> M=L, K in {k0+0..7, k0+16..23}; lanes 16-31 +8
// B (32x16, KxN): lane L<16 -> N=L, K in {k0+0..7, k0+16..23}; lanes 16-31 +8
// C/D f32: VGPR r -> M = r + 8*(lane>>4), N = lane&15
// ---------------------------------------------------------------------------
__device__ __forceinline__ v16bf load_fragA(const bf16_t* __restrict__ base,
                                            int ld, int row0, int k0, int lane) {
  int half = lane >> 4, m = lane & 15;
  const bf16_t* p = base + (size_t)(row0 + m) * ld + k0 + 8 * half;
  FragBits fb;
  fb.lo = *reinterpret_cast<const u32x4*>(p);        // K = +0..7
  fb.hi = *reinterpret_cast<const u32x4*>(p + 16);   // K = +16..23
  return __builtin_bit_cast(v16bf, fb);
}
__device__ __forceinline__ v16bf load_fragB(const bf16_t* __restrict__ base,
                                            int ld, int n0, int k0, int lane) {
  int half = lane >> 4, n = lane & 15;
  const bf16_t* p = base + (size_t)(n0 + n) * ld + k0 + 8 * half;
  FragBits fb;
  fb.lo = *reinterpret_cast<const u32x4*>(p);        // K = +0..7
  fb.hi = *reinterpret_cast<const u32x4*>(p + 16);   // K = +16..23
  return __builtin_bit_cast(v16bf, fb);
}
__device__ __forceinline__ v8f wmma_bf16(v16bf a, v16bf b, v8f c) {
  return __builtin_amdgcn_wmma_f32_16x16x32_bf16(false, a, false, b,
                                                 (short)0, c, false, false);
}

// ---------------------------------------------------------------------------
// f32 -> bf16 conversion, 8 elements/thread (b128 in/out), scalar tail
// ---------------------------------------------------------------------------
__global__ void __launch_bounds__(256)
k_f32_to_bf16(const float* __restrict__ s, bf16_t* __restrict__ d, int n) {
  int stride = gridDim.x * blockDim.x;
  int nv = n >> 3;  // vectors of 8
  for (int i = blockIdx.x * blockDim.x + threadIdx.x; i < nv; i += stride) {
    f32x4 a = *reinterpret_cast<const f32x4*>(s + (size_t)i * 8);
    f32x4 b = *reinterpret_cast<const f32x4*>(s + (size_t)i * 8 + 4);
    u32x4 o;
    o.x = (unsigned)f2bf_bits(a.x) | ((unsigned)f2bf_bits(a.y) << 16);
    o.y = (unsigned)f2bf_bits(a.z) | ((unsigned)f2bf_bits(a.w) << 16);
    o.z = (unsigned)f2bf_bits(b.x) | ((unsigned)f2bf_bits(b.y) << 16);
    o.w = (unsigned)f2bf_bits(b.z) | ((unsigned)f2bf_bits(b.w) << 16);
    *reinterpret_cast<u32x4*>(d + (size_t)i * 8) = o;
  }
  // tail (n not multiple of 8)
  int t0 = nv << 3;
  for (int i = t0 + blockIdx.x * blockDim.x + threadIdx.x; i < n; i += stride)
    d[i] = f2bf(s[i]);
}

// 0.5 * ||codebook row||^2
__global__ void __launch_bounds__(256)
k_cbnorm(const float* __restrict__ cb, float* __restrict__ cnorm, int n, int k) {
  int i = blockIdx.x * blockDim.x + threadIdx.x;
  if (i >= n) return;
  float s = 0.f;
  for (int j = 0; j < k; ++j) { float v = cb[(size_t)i * k + j]; s += v * v; }
  cnorm[i] = 0.5f * s;
}

// ---------------------------------------------------------------------------
// GEMM: out[M,N] = act(A[M,K] @ W[N,K]^T + bias).
// One wave per 16x64 output strip: A fragment loaded once per k-step and
// reused across 4 WMMAs (4 N-tiles). Requires N % 64 == 0.
// ---------------------------------------------------------------------------
__global__ void __launch_bounds__(256)
k_gemm(const bf16_t* __restrict__ A, const bf16_t* __restrict__ W,
       const float* __restrict__ bias, float* __restrict__ outF,
       bf16_t* __restrict__ outB, int M, int N, int K, int relu) {
  int wave = (int)((blockIdx.x * blockDim.x + threadIdx.x) >> 5);
  int lane = threadIdx.x & 31;
  int ntn4 = N >> 6;                       // strips of 4 tiles
  int tm = wave / ntn4, ts = wave - tm * ntn4;
  if (tm >= (M >> 4)) return;
  int n0 = ts << 6;

  v8f acc0 = {}, acc1 = {}, acc2 = {}, acc3 = {};
  for (int k0 = 0; k0 < K; k0 += 32) {
    v16bf a  = load_fragA(A, K, tm << 4, k0, lane);
    v16bf b0 = load_fragB(W, K, n0 +  0, k0, lane);
    v16bf b1 = load_fragB(W, K, n0 + 16, k0, lane);
    v16bf b2 = load_fragB(W, K, n0 + 32, k0, lane);
    v16bf b3 = load_fragB(W, K, n0 + 48, k0, lane);
    acc0 = wmma_bf16(a, b0, acc0);
    acc1 = wmma_bf16(a, b1, acc1);
    acc2 = wmma_bf16(a, b2, acc2);
    acc3 = wmma_bf16(a, b3, acc3);
  }

  int c16  = lane & 15;
  int half = lane >> 4;
  v8f* accs[4] = {&acc0, &acc1, &acc2, &acc3};
#pragma unroll
  for (int j = 0; j < 4; ++j) {
    int n = n0 + (j << 4) + c16;
    float bv = bias ? bias[n] : 0.0f;
    v8f acc = *accs[j];
#pragma unroll
    for (int r = 0; r < 8; ++r) {
      int m = (tm << 4) + r + (half << 3);
      float v = acc[r] + bv;
      if (relu) v = fmaxf(v, 0.0f);
      size_t o = (size_t)m * N + n;
      if (outF) outF[o] = v;
      if (outB) outB[o] = f2bf(v);
    }
  }
}

// ---------------------------------------------------------------------------
// VQ: per 16-row tile of z_e, sweep codebook with WMMA, argmax(z.c - 0.5|c|^2),
// shuffle-reduce across the 16 lanes of each half, gather winning rows.
// ---------------------------------------------------------------------------
__global__ void __launch_bounds__(256)
k_vq(const bf16_t* __restrict__ zeb, const bf16_t* __restrict__ cbB,
     const float* __restrict__ cbF, const float* __restrict__ cnorm,
     int CB, float* __restrict__ zq, bf16_t* __restrict__ zqb, int M) {
  int wave = (int)((blockIdx.x * blockDim.x + threadIdx.x) >> 5);
  int lane = threadIdx.x & 31;
  if (wave >= (M >> 4)) return;
  int half = lane >> 4, c16 = lane & 15;

  v16bf a0 = load_fragA(zeb, 64, wave << 4, 0, lane);
  v16bf a1 = load_fragA(zeb, 64, wave << 4, 32, lane);

  float best[8]; int bidx[8];
#pragma unroll
  for (int r = 0; r < 8; ++r) { best[r] = -3.4e38f; bidx[r] = 0; }

  for (int t = 0; t < (CB >> 4); ++t) {
    v16bf b0 = load_fragB(cbB, 64, t << 4, 0, lane);
    v16bf b1 = load_fragB(cbB, 64, t << 4, 32, lane);
    v8f acc = {};
    acc = wmma_bf16(a0, b0, acc);
    acc = wmma_bf16(a1, b1, acc);
    int col  = (t << 4) + c16;
    float cn = cnorm[col];
#pragma unroll
    for (int r = 0; r < 8; ++r) {
      float s = acc[r] - cn;
      if (s > best[r]) { best[r] = s; bidx[r] = col; }
    }
  }

#pragma unroll
  for (int r = 0; r < 8; ++r) {
    // reduce over the 16 lanes of this half (xor masks stay inside the group)
    for (int mk = 1; mk < 16; mk <<= 1) {
      float ov = __shfl_xor(best[r], mk, 32);
      int   oi = __shfl_xor(bidx[r], mk, 32);
      if (ov > best[r] || (ov == best[r] && oi < bidx[r])) { best[r] = ov; bidx[r] = oi; }
    }
    int row = (wave << 4) + r + (half << 3);
    int idx = bidx[r];
    // 16 lanes of this half each copy 4 consecutive codebook floats
    f32x4 cv = *reinterpret_cast<const f32x4*>(cbF + (size_t)idx * 64 + (c16 << 2));
    *reinterpret_cast<f32x4*>(zq + (size_t)row * 64 + (c16 << 2)) = cv;
    u32x2 pk;
    pk.x = (unsigned)f2bf_bits(cv.x) | ((unsigned)f2bf_bits(cv.y) << 16);
    pk.y = (unsigned)f2bf_bits(cv.z) | ((unsigned)f2bf_bits(cv.w) << 16);
    *reinterpret_cast<u32x2*>(zqb + (size_t)row * 64 + (c16 << 2)) = pk;
  }
}

// ---------------------------------------------------------------------------
// Loss partial sums: commit = sum (ze-zq)^2 ; vq = sum (zq - (ze+att))^2
// ---------------------------------------------------------------------------
__global__ void __launch_bounds__(256)
k_loss(const float* __restrict__ ze, const float* __restrict__ zq,
       const float* __restrict__ att, int n,
       float* __restrict__ pc, float* __restrict__ pv) {
  __shared__ float sc[256], sv[256];
  int tid = threadIdx.x;
  float c = 0.f, v = 0.f;
  int stride = gridDim.x * blockDim.x;
  for (int i = blockIdx.x * blockDim.x + tid; i < n; i += stride) {
    float e = ze[i], q = zq[i];
    float d = e - q;  c += d * d;
    float a = att ? att[i] : 0.f;
    float w = q - (e + a);  v += w * w;
  }
  sc[tid] = c; sv[tid] = v;
  __syncthreads();
  for (int s = 128; s > 0; s >>= 1) {
    if (tid < s) { sc[tid] += sc[tid + s]; sv[tid] += sv[tid + s]; }
    __syncthreads();
  }
  if (tid == 0) { pc[blockIdx.x] = sc[0]; pv[blockIdx.x] = sv[0]; }
}

__global__ void __launch_bounds__(256)
k_finalize(const float* __restrict__ parts, int nb, float* __restrict__ out,
           float invn) {
  __shared__ float sc[256], sv[256];
  int tid = threadIdx.x;
  float c = 0.f, v = 0.f;
  for (int l = 0; l < 3; ++l) {
    const float* pc = parts + (size_t)l * 2 * nb;
    const float* pv = pc + nb;
    for (int i = tid; i < nb; i += blockDim.x) { c += pc[i]; v += pv[i]; }
  }
  sc[tid] = c; sv[tid] = v;
  __syncthreads();
  for (int s = 128; s > 0; s >>= 1) {
    if (tid < s) { sc[tid] += sc[tid + s]; sv[tid] += sv[tid + s]; }
    __syncthreads();
  }
  if (tid == 0) { out[0] = sv[0] * invn; out[1] = sc[0] * invn; }
}

// ---------------------------------------------------------------------------
// Host orchestration
// ---------------------------------------------------------------------------
extern "C" void kernel_launch(void* const* d_in, const int* in_sizes, int n_in,
                              void* d_out, int out_size, void* d_ws, size_t ws_size,
                              hipStream_t stream) {
  constexpr int B = 16384, IN = 512, HID = 256, LAT = 64;
  constexpr int NB = 512;  // loss reduction blocks

  char* ws = (char*)d_ws;
  size_t off = 0;
  auto alloc = [&](size_t bytes) -> void* {
    off = (off + 255) & ~(size_t)255;
    void* p = ws + off; off += bytes; return p;
  };

  bf16_t* xb   = (bf16_t*)alloc((size_t)B * IN  * 2);
  bf16_t* h1   = (bf16_t*)alloc((size_t)B * HID * 2);
  bf16_t* h2   = (bf16_t*)alloc((size_t)B * HID * 2);
  bf16_t* zeb  = (bf16_t*)alloc((size_t)B * LAT * 2);
  bf16_t* zqbA = (bf16_t*)alloc((size_t)B * LAT * 2);
  bf16_t* zqbB = (bf16_t*)alloc((size_t)B * LAT * 2);
  bf16_t* vb   = (bf16_t*)alloc((size_t)B * LAT * 2);
  bf16_t* wb   = (bf16_t*)alloc((size_t)HID * IN * 2);   // largest weight 256x512
  bf16_t* cbB  = (bf16_t*)alloc((size_t)4096 * LAT * 2);
  float*  cnorm= (float*) alloc((size_t)4096 * 4);
  float*  ze   = (float*) alloc((size_t)B * LAT * 4);
  float*  zq   = (float*) alloc((size_t)B * LAT * 4);
  float*  att  = (float*) alloc((size_t)B * LAT * 4);
  float*  parts= (float*) alloc((size_t)3 * 2 * NB * 4);

  auto conv = [&](const float* s, bf16_t* d, int n) {
    int grid = (n / 8 + 256 * 4 - 1) / (256 * 4);
    if (grid < 1) grid = 1;
    k_f32_to_bf16<<<grid, 256, 0, stream>>>(s, d, n);
  };
  auto gemm = [&](const bf16_t* A, const bf16_t* Wm, const float* bias,
                  float* oF, bf16_t* oB, int N, int K, int relu) {
    int waves = (B >> 4) * (N >> 6);       // one wave per 16x64 strip
    k_gemm<<<waves / 8, 256, 0, stream>>>(A, Wm, bias, oF, oB, B, N, K, relu);
  };

  float* outF = (float*)d_out;
  bf16_t* zq_prev_b = nullptr;

  for (int l = 0; l < 3; ++l) {
    const int base = 3 + 13 * l;
    const float* x      = (const float*)d_in[l];
    const float* ew1    = (const float*)d_in[base + 0];
    const float* eb1    = (const float*)d_in[base + 1];
    const float* ew2    = (const float*)d_in[base + 2];
    const float* eb2    = (const float*)d_in[base + 3];
    const float* ew3    = (const float*)d_in[base + 4];
    const float* eb3    = (const float*)d_in[base + 5];
    const float* dw1    = (const float*)d_in[base + 6];
    const float* db1    = (const float*)d_in[base + 7];
    const float* dw2    = (const float*)d_in[base + 8];
    const float* db2    = (const float*)d_in[base + 9];
    const float* dw3    = (const float*)d_in[base + 10];
    const float* db3    = (const float*)d_in[base + 11];
    const float* cbF    = (const float*)d_in[base + 12];
    const int CB        = in_sizes[base + 12] / LAT;

    // ---- encoder ----
    conv(x, xb, B * IN);
    conv(ew1, wb, HID * IN);   gemm(xb, wb, eb1, nullptr, h1, HID, IN, 1);
    conv(ew2, wb, HID * HID);  gemm(h1, wb, eb2, nullptr, h2, HID, HID, 1);
    conv(ew3, wb, LAT * HID);  gemm(h2, wb, eb3, ze, zeb, LAT, HID, 0);

    // ---- attention (l>0): att = (zq_prev @ wv^T + bv) @ ow^T + ob ----
    const float* attp = nullptr;
    if (l > 0) {
      const int ab = 42 + 4 * (l - 1);
      const float* in_w  = (const float*)d_in[ab + 0];
      const float* in_b  = (const float*)d_in[ab + 1];
      const float* out_w = (const float*)d_in[ab + 2];
      const float* out_b = (const float*)d_in[ab + 3];
      conv(in_w + 2 * LAT * LAT, wb, LAT * LAT);           // wv = in_w[128:192]
      gemm(zq_prev_b, wb, in_b + 2 * LAT, nullptr, vb, LAT, LAT, 0);
      conv(out_w, wb, LAT * LAT);
      gemm(vb, wb, out_b, att, nullptr, LAT, LAT, 0);
      attp = att;
    }

    // ---- quantize ----
    conv(cbF, cbB, CB * LAT);
    k_cbnorm<<<(CB + 255) / 256, 256, 0, stream>>>(cbF, cnorm, CB, LAT);
    bf16_t* zq_cur_b = (l & 1) ? zqbB : zqbA;
    k_vq<<<(B >> 4) / 8, 256, 0, stream>>>(zeb, cbB, cbF, cnorm, CB, zq, zq_cur_b, B);

    // ---- losses (partial sums) ----
    k_loss<<<NB, 256, 0, stream>>>(ze, zq, attp, B * LAT,
                                   parts + (size_t)l * 2 * NB,
                                   parts + (size_t)l * 2 * NB + NB);

    // ---- decoder -> recon into d_out ----
    conv(dw1, wb, HID * LAT);  gemm(zq_cur_b, wb, db1, nullptr, h1, HID, LAT, 1);
    conv(dw2, wb, HID * HID);  gemm(h1, wb, db2, nullptr, h2, HID, HID, 1);
    conv(dw3, wb, IN * HID);   gemm(h2, wb, db3, outF + (size_t)l * B * IN, nullptr, IN, HID, 0);

    zq_prev_b = zq_cur_b;
  }

  // scalars: [vq_loss, commit_loss] right after the 3 recon tensors
  k_finalize<<<1, 256, 0, stream>>>(parts, NB, outF + (size_t)3 * B * IN,
                                    1.0f / (float)(B * LAT));
}
